// DeformableConv2d_37357625540688
// MI455X (gfx1250) — matrix operand order repaired and checked by reading, not compiled, not run on hardware
//
#include <hip/hip_runtime.h>
#include <hip/hip_bf16.h>

typedef __attribute__((ext_vector_type(16))) _Float16 v16h;
typedef __attribute__((ext_vector_type(8)))  float    v8f;
typedef __attribute__((ext_vector_type(4)))  float    v4f;

static constexpr int Bn = 4, C = 64, H = 128, Wd = 128, OC = 64;
static constexpr int HW = H * Wd;                // 16384
static constexpr int TILES = Bn * HW / 16;       // 4096 (16 pixels per wave-tile)

// ---------------------------------------------------------------------------
// Pass 0a: transpose NCHW -> NHWC (coalesced both ways via padded LDS tile).
// Used for x, offset_map and modulator_map.
// ---------------------------------------------------------------------------
__global__ void k_transpose(const float* __restrict__ x, float* __restrict__ xt) {
    __shared__ float tile[64 * 129];
    int by = blockIdx.x;                 // b*H + y
    int b = by / H, y = by % H;
    const float* src = x + (size_t)b * C * HW + (size_t)y * Wd;
    for (int i = threadIdx.x; i < 64 * 128; i += blockDim.x) {
        int c = i >> 7, xx = i & 127;
        tile[c * 129 + xx] = src[(size_t)c * HW + xx];
    }
    __syncthreads();
    float* dst = xt + (size_t)by * Wd * 64;
    for (int i = threadIdx.x; i < 64 * 128; i += blockDim.x) {
        int xx = i >> 6, c = i & 63;
        dst[(size_t)xx * 64 + c] = tile[c * 129 + xx];
    }
}

// ---------------------------------------------------------------------------
// Pass 0b: pre-pack all three weight tensors into WMMA B-fragment layout f16.
//   BW : main weight (64,64,3,3)  -> 18 chunks x 4 N-tiles
//   OWp: offset_w (18,64,3,3), N padded to 32 -> 18 chunks x 2 N-tiles
//   MWp: mod_w    ( 9,64,3,3), N padded to 16 -> 18 chunks x 1 N-tile
// Fragment layout (ISA 7.12.2, B 32x16 f16): lane n=L&15, g=L>>4;
// half h -> K = 8g+h (h<8), 16+8g+(h-8) (h>=8) -> channel c = c0 + K.
// ---------------------------------------------------------------------------
__global__ void k_prepack(const float* __restrict__ w,
                          const float* __restrict__ ow,
                          const float* __restrict__ mw,
                          _Float16* __restrict__ BW,
                          _Float16* __restrict__ OWp,
                          _Float16* __restrict__ MWp) {
    int i = blockIdx.x * blockDim.x + threadIdx.x;
    if (i < 36864) {                                  // BW: 18*4*32*16
        int h = i & 15, L = (i >> 4) & 31, t = (i >> 9) & 3, j = i >> 11;
        int k2 = j >> 1, c0 = (j & 1) * 32;
        int n = L & 15, g = L >> 4;
        int c = c0 + g * 8 + h + (h >= 8 ? 8 : 0), o = t * 16 + n;
        BW[i] = (_Float16)w[(o * 64 + c) * 9 + k2];
    } else if (i < 36864 + 18432) {                   // OWp: 18*2*32*16
        int ii = i - 36864;
        int h = ii & 15, L = (ii >> 4) & 31, t = (ii >> 9) & 1, j = ii >> 10;
        int k2 = j >> 1, c0 = (j & 1) * 32;
        int n = L & 15, g = L >> 4;
        int c = c0 + g * 8 + h + (h >= 8 ? 8 : 0), o = t * 16 + n;
        OWp[ii] = (o < 18) ? (_Float16)ow[(o * 64 + c) * 9 + k2] : (_Float16)0.f;
    } else if (i < 36864 + 18432 + 9216) {            // MWp: 18*1*32*16
        int ii = i - 36864 - 18432;
        int h = ii & 15, L = (ii >> 4) & 31, j = ii >> 9;
        int k2 = j >> 1, c0 = (j & 1) * 32;
        int n = L & 15, g = L >> 4;
        int c = c0 + g * 8 + h + (h >= 8 ? 8 : 0), o = n;
        MWp[ii] = (o < 9) ? (_Float16)mw[(o * 64 + c) * 9 + k2] : (_Float16)0.f;
    }
}

// ---------------------------------------------------------------------------
// Pass 1 (WMMA): offset conv (18ch, pad32) + mask conv (9ch, pad16) as GEMMs.
// One wave = 16 pixels; K = 576 in 18 chunks; 3 WMMA per chunk.
// Conv results go through LDS, then per-(pixel,k2) bilinear state is emitted.
// ---------------------------------------------------------------------------
__global__ void __launch_bounds__(256) k_offmask_wmma(
        const float* __restrict__ omt, const float* __restrict__ mmt,
        const _Float16* __restrict__ OWp, const _Float16* __restrict__ MWp,
        const float* __restrict__ ob, const float* __restrict__ mb,
        int* __restrict__ Y0, int* __restrict__ X0,
        float* __restrict__ WY, float* __restrict__ WX, float* __restrict__ MV) {
    __shared__ float sm[8][16 * 49];
    int wave = threadIdx.x >> 5, lane = threadIdx.x & 31;
    int tile = blockIdx.x * 8 + wave;
    int b = tile >> 10, r = tile & 1023;
    int y = r >> 3, x0 = (r & 7) << 4;
    int g = lane >> 4, n = lane & 15;
    int xm = x0 + n;                                  // this lane's pixel (A rows)

    v8f ao0 = {0}, ao1 = {0}, am = {0};
    const float* ombt = omt + (size_t)b * HW * 64;
    const float* mmbt = mmt + (size_t)b * HW * 64;

    for (int j = 0; j < 18; j++) {
        int k2 = j >> 1, c0 = (j & 1) << 5;
        int yy = y - 1 + k2 / 3;
        int xx = xm - 1 + k2 % 3;
        bool valid = (yy >= 0) && (yy < H) && (xx >= 0) && (xx < Wd);
        float vf = valid ? 1.f : 0.f;
        int yi = min(max(yy, 0), H - 1), xi = min(max(xx, 0), Wd - 1);
        size_t sp = ((size_t)yi * Wd + xi) * 64;
        int cr0 = c0 + 8 * g, cr1 = cr0 + 16;

        v4f o0 = *(const v4f*)(ombt + sp + cr0);
        v4f o1 = *(const v4f*)(ombt + sp + cr0 + 4);
        v4f o2 = *(const v4f*)(ombt + sp + cr1);
        v4f o3 = *(const v4f*)(ombt + sp + cr1 + 4);
        v4f q0 = *(const v4f*)(mmbt + sp + cr0);
        v4f q1 = *(const v4f*)(mmbt + sp + cr0 + 4);
        v4f q2 = *(const v4f*)(mmbt + sp + cr1);
        v4f q3 = *(const v4f*)(mmbt + sp + cr1 + 4);

        v16h aoff, amask;
        #pragma unroll
        for (int q = 0; q < 4; q++) {
            aoff[q]       = (_Float16)(vf * o0[q]);
            aoff[4 + q]   = (_Float16)(vf * o1[q]);
            aoff[8 + q]   = (_Float16)(vf * o2[q]);
            aoff[12 + q]  = (_Float16)(vf * o3[q]);
            amask[q]      = (_Float16)(vf * q0[q]);
            amask[4 + q]  = (_Float16)(vf * q1[q]);
            amask[8 + q]  = (_Float16)(vf * q2[q]);
            amask[12 + q] = (_Float16)(vf * q3[q]);
        }

        const v16h* bwo = (const v16h*)OWp + (size_t)(j * 2) * 32 + lane;
        v16h bo0 = bwo[0];
        v16h bo1 = bwo[32];
        v16h bmv = ((const v16h*)MWp)[(size_t)j * 32 + lane];

        ao0 = __builtin_amdgcn_wmma_f32_16x16x32_f16(false, aoff,  false, bo0, (short)0, ao0, false, false);
        ao1 = __builtin_amdgcn_wmma_f32_16x16x32_f16(false, aoff,  false, bo1, (short)0, ao1, false, false);
        am  = __builtin_amdgcn_wmma_f32_16x16x32_f16(false, amask, false, bmv, (short)0, am,  false, false);
    }

    // D layout: N = lane&15, M = rr + 8*(lane>>4) -> stage into LDS
    float* sw = sm[wave];
    #pragma unroll
    for (int rr = 0; rr < 8; rr++) {
        int px = rr + 8 * g;
        sw[px * 49 + n]      = ao0[rr];   // offset channels 0..15
        sw[px * 49 + 16 + n] = ao1[rr];   // offset channels 16..31 (only 16,17 used)
        sw[px * 49 + 32 + n] = am[rr];    // mask channels 0..15   (only 0..8 used)
    }
    __syncthreads();

    for (int i = lane; i < 144; i += 32) {            // 16 px x 9 k2
        int px = i / 9, k2 = i - px * 9;
        float dyv = sw[px * 49 + 2 * k2]     + ob[2 * k2];
        float dxv = sw[px * 49 + 2 * k2 + 1] + ob[2 * k2 + 1];
        float tv  = sw[px * 49 + 32 + k2]    + mb[k2];
        float mval = 2.f / (1.f + expf(-tv));
        float py  = dyv + (float)(y - 1 + k2 / 3);
        float pxf = dxv + (float)(x0 + px - 1 + k2 % 3);
        float fy = floorf(py), fx = floorf(pxf);
        size_t idx = (((size_t)b * 9 + k2) * H + y) * Wd + x0 + px;
        Y0[idx] = (int)fy; X0[idx] = (int)fx;
        WY[idx] = py - fy;  WX[idx] = pxf - fx;  MV[idx] = mval;
    }
}

// ---------------------------------------------------------------------------
// Pass 2: deformable GEMM. One wave = 16 pixels x 64 outputs.
// ---------------------------------------------------------------------------
__global__ void __launch_bounds__(256) k_deform_gemm(
        const float* __restrict__ xt, const _Float16* __restrict__ BW,
        const int* __restrict__ Y0, const int* __restrict__ X0,
        const float* __restrict__ WY, const float* __restrict__ WX,
        const float* __restrict__ MV, float* __restrict__ out) {
    __shared__ float sm[8][16 * 65];
    int wave = threadIdx.x >> 5;
    int lane = threadIdx.x & 31;
    int tile = blockIdx.x * 8 + wave;
    int b = tile >> 10;
    int r = tile & 1023;
    int y = r >> 3;
    int x0 = (r & 7) << 4;
    int g = lane >> 4, m = lane & 15;
    int xm = x0 + m;

    v8f acc0 = {0}, acc1 = {0}, acc2 = {0}, acc3 = {0};
    size_t ibase = (((size_t)b * 9) * H + y) * Wd + xm;   // + k2*HW
    const float* xb = xt + (size_t)b * HW * 64;

    for (int j = 0; j < 18; j++) {
        int k2 = j >> 1;
        int c0 = (j & 1) << 5;
        size_t ii = ibase + (size_t)k2 * HW;
        int y0 = Y0[ii], xi0 = X0[ii];
        float wy = WY[ii], wx = WX[ii], mv = MV[ii];
        float w00 = (1.f - wy) * (1.f - wx) * mv;
        float w01 = (1.f - wy) * wx * mv;
        float w10 = wy * (1.f - wx) * mv;
        float w11 = wy * wx * mv;

        float s[16];
        #pragma unroll
        for (int q = 0; q < 16; q++) s[q] = 0.f;
        int cr0 = c0 + 8 * g;        // A half h<8 : channels c0+8g .. +7 (K=8g+h)
        int cr1 = cr0 + 16;          // A half h>=8: channels c0+16+8g .. +7

        #pragma unroll
        for (int corner = 0; corner < 4; corner++) {
            int iy = corner >> 1, ix = corner & 1;
            int yc = y0 + iy, xc = xi0 + ix;
            float wc = (corner == 0) ? w00 : (corner == 1) ? w01 : (corner == 2) ? w10 : w11;
            bool valid = (yc >= 0) && (yc < H) && (xc >= 0) && (xc < Wd);
            wc = valid ? wc : 0.f;
            int yi = min(max(yc, 0), H - 1), xi = min(max(xc, 0), Wd - 1);
            const float* p = xb + ((size_t)yi * Wd + xi) * 64;
            v4f a0 = *(const v4f*)(p + cr0);
            v4f a1 = *(const v4f*)(p + cr0 + 4);
            v4f b0 = *(const v4f*)(p + cr1);
            v4f b1 = *(const v4f*)(p + cr1 + 4);
            #pragma unroll
            for (int q = 0; q < 4; q++) {
                s[q]      = fmaf(wc, a0[q], s[q]);
                s[4 + q]  = fmaf(wc, a1[q], s[4 + q]);
                s[8 + q]  = fmaf(wc, b0[q], s[8 + q]);
                s[12 + q] = fmaf(wc, b1[q], s[12 + q]);
            }
        }

        v16h a;
        #pragma unroll
        for (int q = 0; q < 16; q++) a[q] = (_Float16)s[q];

        const v16h* bw = (const v16h*)BW + (size_t)(j * 4) * 32 + lane;
        v16h bf0 = bw[0];
        v16h bf1 = bw[32];
        v16h bf2 = bw[64];
        v16h bf3 = bw[96];

        acc0 = __builtin_amdgcn_wmma_f32_16x16x32_f16(false, a, false, bf0, (short)0, acc0, false, false);
        acc1 = __builtin_amdgcn_wmma_f32_16x16x32_f16(false, a, false, bf1, (short)0, acc1, false, false);
        acc2 = __builtin_amdgcn_wmma_f32_16x16x32_f16(false, a, false, bf2, (short)0, acc2, false, false);
        acc3 = __builtin_amdgcn_wmma_f32_16x16x32_f16(false, a, false, bf3, (short)0, acc3, false, false);
    }

    // Epilogue: D layout (N = lane%16, M = r+8*(lane/16)) -> LDS -> coalesced stores
    float* sw = sm[wave];
    #pragma unroll
    for (int rr = 0; rr < 8; rr++) {
        int px = rr + 8 * g;
        sw[px * 65 + m +  0] = acc0[rr];
        sw[px * 65 + m + 16] = acc1[rr];
        sw[px * 65 + m + 32] = acc2[rr];
        sw[px * 65 + m + 48] = acc3[rr];
    }
    __syncthreads();
    float* ob = out + (size_t)b * OC * HW + (size_t)y * Wd + x0;
    #pragma unroll
    for (int i = 0; i < 32; i++) {
        int o = 2 * i + g;
        ob[(size_t)o * HW + m] = sw[m * 65 + o];
    }
}

// ---------------------------------------------------------------------------
extern "C" void kernel_launch(void* const* d_in, const int* in_sizes, int n_in,
                              void* d_out, int out_size, void* d_ws, size_t ws_size,
                              hipStream_t stream) {
    const float* x        = (const float*)d_in[0];
    const float* om       = (const float*)d_in[1];
    const float* mm       = (const float*)d_in[2];
    const float* offset_w = (const float*)d_in[3];
    const float* offset_b = (const float*)d_in[4];
    const float* mod_w    = (const float*)d_in[5];
    const float* mod_b    = (const float*)d_in[6];
    const float* weight   = (const float*)d_in[7];
    float* out = (float*)d_out;

    char* ws = (char*)d_ws;
    const size_t T_BYTES   = (size_t)Bn * HW * 64 * 4;       // 16 MB per transposed map
    const size_t ARR_BYTES = (size_t)Bn * 9 * HW * 4;        // 2.25 MB each
    float*     xt  = (float*)(ws);
    float*     omt = (float*)(ws + T_BYTES);
    float*     mmt = (float*)(ws + 2 * T_BYTES);
    char*      p   = ws + 3 * T_BYTES;
    int*       Y0  = (int*)  (p);
    int*       X0  = (int*)  (p + ARR_BYTES);
    float*     WY  = (float*)(p + 2 * ARR_BYTES);
    float*     WX  = (float*)(p + 3 * ARR_BYTES);
    float*     MV  = (float*)(p + 4 * ARR_BYTES);
    _Float16*  BW  = (_Float16*)(p + 5 * ARR_BYTES);
    _Float16*  OWp = BW + 36864;
    _Float16*  MWp = OWp + 18432;

    k_transpose<<<Bn * H, 256, 0, stream>>>(x, xt);
    k_transpose<<<Bn * H, 256, 0, stream>>>(om, omt);
    k_transpose<<<Bn * H, 256, 0, stream>>>(mm, mmt);
    k_prepack<<<(36864 + 18432 + 9216 + 255) / 256, 256, 0, stream>>>(
        weight, offset_w, mod_w, BW, OWp, MWp);
    k_offmask_wmma<<<TILES / 8, 256, 0, stream>>>(omt, mmt, OWp, MWp, offset_b, mod_b,
                                                  Y0, X0, WY, WX, MV);
    k_deform_gemm<<<TILES / 8, 256, 0, stream>>>(xt, BW, Y0, X0, WY, WX, MV, out);
}
